// MultiHeadAttention_52218212384825
// MI455X (gfx1250) — matrix operand order repaired and checked by reading, not compiled, not run on hardware
//
#include <hip/hip_runtime.h>
#include <hip/hip_bf16.h>

typedef _Float16 f16;
typedef _Float16 v16h __attribute__((ext_vector_type(16)));
typedef _Float16 v8h  __attribute__((ext_vector_type(8)));
typedef float    v8f  __attribute__((ext_vector_type(8)));
typedef int      v4i  __attribute__((ext_vector_type(4)));

typedef __attribute__((address_space(1))) v4i gv4i;   // global int4
typedef __attribute__((address_space(3))) v4i sv4i;   // LDS int4

#define NH 12
#define HD 64
#define NE 768
#define TT 2048
#define BB 4

#if defined(__gfx1250__) && __has_builtin(__builtin_amdgcn_global_load_async_to_lds_b128)
#define USE_ASYNC_LDS 1
#else
#define USE_ASYNC_LDS 0
#endif

__device__ __forceinline__ void cp16_to_lds(f16* lds, const f16* g) {
#if USE_ASYNC_LDS
  // GLOBAL_LOAD_ASYNC_TO_LDS_B128: per-lane 16B global -> LDS, tracked by ASYNCcnt.
  __builtin_amdgcn_global_load_async_to_lds_b128(
      (gv4i*)g, (sv4i*)lds, 0, 0);
#else
  *(v8h*)lds = *(const v8h*)g;
#endif
}

__device__ __forceinline__ void cp_async_wait() {
#if USE_ASYNC_LDS
#if __has_builtin(__builtin_amdgcn_s_wait_asynccnt)
  __builtin_amdgcn_s_wait_asynccnt(0);
#else
  asm volatile("s_wait_asynccnt 0x0" ::: "memory");
#endif
#endif
}

__device__ __forceinline__ v8f vzero() {
  v8f z = {0.f,0.f,0.f,0.f,0.f,0.f,0.f,0.f};
  return z;
}

__device__ __forceinline__ v16h cat8(v8h lo, v8h hi) {
  return __builtin_shufflevector(lo, hi, 0,1,2,3,4,5,6,7,8,9,10,11,12,13,14,15);
}

__device__ __forceinline__ v8f wmma_f16(v16h a, v16h b, v8f c) {
  // D = A(16x32 f16) * B(32x16 f16) + C(16x16 f32)
  return __builtin_amdgcn_wmma_f32_16x16x32_f16(false, a, false, b, (short)0, c, false, false);
}

// ---------------------------------------------------------------------------
// Kernel 0: one-time x fp32 -> f16 conversion (read 1x instead of 36x fp32).
// ---------------------------------------------------------------------------
__global__ __launch_bounds__(256) void cvt_x_kernel(const float* __restrict__ x,
                                                    f16* __restrict__ xh) {
  size_t i = ((size_t)blockIdx.x * 256 + threadIdx.x) * 8;
  float4 a = *(const float4*)&x[i];
  float4 b = *(const float4*)&x[i + 4];
  v8h o;
  o[0]=(f16)a.x; o[1]=(f16)a.y; o[2]=(f16)a.z; o[3]=(f16)a.w;
  o[4]=(f16)b.x; o[5]=(f16)b.y; o[6]=(f16)b.z; o[7]=(f16)b.w;
  *(v8h*)&xh[i] = o;
}

// ---------------------------------------------------------------------------
// Kernel 1: per-head QKV projections.
//   xh [B*T, 768] f16  x  W[head] [768, 64] fp32 -> f16 outputs.
//   Q pre-scaled by 768^-0.5; V written transposed [B,H,D,T].
// Block: 256 threads (8 waves). Tile: 256 (M) x 64 (N), K-step 32.
// Each wave owns 32 rows (2 A fragments) so every B fragment feeds 2 WMMAs.
// ---------------------------------------------------------------------------
__global__ __launch_bounds__(256) void qkv_proj_kernel(
    const f16* __restrict__ xh,
    const float* __restrict__ Wq,
    const float* __restrict__ Wk,
    const float* __restrict__ Wv,
    f16* __restrict__ qbuf,   // [B,H,T,64]
    f16* __restrict__ kbuf,   // [B,H,T,64]
    f16* __restrict__ vtbuf)  // [B,H,64,T]
{
  __shared__ f16 xt[256 * 32];   // A tile, row-major [m][k]   (16 KB)
  __shared__ f16 wt[64 * 32];    // B tile transposed [n(d)][k(c)] (4 KB)

  const int tid   = threadIdx.x;
  const int m0    = blockIdx.x * 256;
  const int head  = blockIdx.y / 3;
  const int which = blockIdx.y % 3;
  const float* W = (which == 0 ? Wq : (which == 1 ? Wk : Wv)) + (size_t)head * NE * HD;

  const int wave = tid >> 5;
  const int lane = tid & 31;
  const int lm   = lane & 15;
  const int h    = lane >> 4;

  v8f acc[2][4];
  #pragma unroll
  for (int s = 0; s < 2; ++s)
    #pragma unroll
    for (int i = 0; i < 4; ++i) acc[s][i] = vzero();

  for (int kc = 0; kc < NE; kc += 32) {
    // Stage A tile 256x32 f16 via async 16B copies (1024 chunks / 256 threads).
    #pragma unroll
    for (int it = 0; it < 4; ++it) {
      int f   = tid + it * 256;        // 0..1023
      int row = f >> 2;                // 4 chunks of 8 per row
      int c8  = (f & 3) << 3;
      cp16_to_lds(&xt[row * 32 + c8], &xh[(size_t)(m0 + row) * NE + kc + c8]);
    }
    // Stage W tile 32x64 transposed into wt[d][c] (fp32 -> f16).
    #pragma unroll
    for (int it = 0; it < 2; ++it) {
      int f  = tid + it * 256;         // 0..511 float4
      int c  = f >> 4;                 // 16 float4 per row of 64
      int d4 = (f & 15) << 2;
      float4 v = *(const float4*)&W[(size_t)(kc + c) * HD + d4];
      wt[(d4 + 0) * 32 + c] = (f16)v.x;
      wt[(d4 + 1) * 32 + c] = (f16)v.y;
      wt[(d4 + 2) * 32 + c] = (f16)v.z;
      wt[(d4 + 3) * 32 + c] = (f16)v.w;
    }
    cp_async_wait();
    __syncthreads();

    // Two A fragments per wave (rows wave*32 .. wave*32+31).
    const int r0 = wave * 32 + lm;
    v16h a0 = cat8(*(const v8h*)&xt[r0 * 32 + h * 16],
                   *(const v8h*)&xt[r0 * 32 + h * 16 + 8]);
    v16h a1 = cat8(*(const v8h*)&xt[(r0 + 16) * 32 + h * 16],
                   *(const v8h*)&xt[(r0 + 16) * 32 + h * 16 + 8]);

    #pragma unroll
    for (int nt = 0; nt < 4; ++nt) {
      v16h bf = *(const v16h*)&wt[(nt * 16 + lm) * 32 + h * 16];
      acc[0][nt] = wmma_f16(a0, bf, acc[0][nt]);
      acc[1][nt] = wmma_f16(a1, bf, acc[1][nt]);
    }
    __syncthreads();
  }

  // Epilogue: branch is block-uniform -> hoisted out of the store loops.
  if (which == 2) {
    #pragma unroll
    for (int s = 0; s < 2; ++s) {
      int g0 = m0 + wave * 32 + s * 16;     // 16-row band, never crosses batch
      int b  = g0 >> 11;
      int t0 = g0 & (TT - 1);
      size_t bh = (size_t)(b * NH + head);
      #pragma unroll
      for (int nt = 0; nt < 4; ++nt) {
        int d = nt * 16 + lm;
        #pragma unroll
        for (int r = 0; r < 8; ++r)
          vtbuf[(bh * HD + d) * TT + t0 + r + 8 * h] = (f16)acc[s][nt][r];
      }
    }
  } else {
    f16* dst = (which == 0) ? qbuf : kbuf;
    const float scale = (which == 0) ? 0.036084391824351615f : 1.0f;  // 768^-0.5
    #pragma unroll
    for (int s = 0; s < 2; ++s) {
      int g0 = m0 + wave * 32 + s * 16;
      int b  = g0 >> 11;
      int t0 = g0 & (TT - 1);
      size_t bh = (size_t)(b * NH + head);
      #pragma unroll
      for (int nt = 0; nt < 4; ++nt) {
        int d = nt * 16 + lm;
        #pragma unroll
        for (int r = 0; r < 8; ++r)
          dst[(bh * TT + t0 + r + 8 * h) * HD + d] = (f16)(acc[s][nt][r] * scale);
      }
    }
  }
}

// ---------------------------------------------------------------------------
// Kernel 2: causal flash attention, one (b, head, 128-query tile) per block.
// Each wave owns 16 queries. We compute S^T = K * Q^T so the exp'd
// accumulator is already in A-fragment layout for P*V (no transpose).
// Online softmax, fp32 accumulation, no LDS needed (waves independent).
// ---------------------------------------------------------------------------
__global__ __launch_bounds__(256) void attn_kernel(
    const f16* __restrict__ qbuf,
    const f16* __restrict__ kbuf,
    const f16* __restrict__ vtbuf,
    f16* __restrict__ att)   // [B, T, 768] (heads concatenated)
{
  const int tid  = threadIdx.x;
  const int wave = tid >> 5;
  const int lane = tid & 31;
  const int lm   = lane & 15;
  const int h    = lane >> 4;
  const int head = blockIdx.y;
  const int b    = blockIdx.z;
  const int q0   = blockIdx.x * 128 + wave * 16;

  const size_t bh = (size_t)(b * NH + head);
  const f16* qh = qbuf  + bh * TT * HD;
  const f16* kh = kbuf  + bh * TT * HD;
  const f16* vh = vtbuf + bh * HD * TT;

  // Resident Q^T B-fragments: lane holds query col N=lane%16, elements d = ks*32 + h*16 + e.
  v16h Qb[2];
  Qb[0] = *(const v16h*)&qh[(size_t)(q0 + lm) * HD + 0 * 32 + h * 16];
  Qb[1] = *(const v16h*)&qh[(size_t)(q0 + lm) * HD + 1 * 32 + h * 16];

  v8f o[4];
  #pragma unroll
  for (int i = 0; i < 4; ++i) o[i] = vzero();
  float mstat = -1e30f;
  float lstat = 0.f;

  for (int kblk = 0; kblk <= q0 + 15; kblk += 32) {
    if (kblk + 32 < TT)
      __builtin_prefetch(&kh[(size_t)(kblk + 32 + lm) * HD], 0, 0);

    // S^T tiles: st0 = keys [kblk, kblk+16), st1 = keys [kblk+16, kblk+32).
    v8f st0 = vzero(), st1 = vzero();
    #pragma unroll
    for (int ks = 0; ks < 2; ++ks) {
      {
        const f16* kr = &kh[(size_t)(kblk + lm) * HD + ks * 32];
        v16h a = cat8(*(const v8h*)&kr[h * 16], *(const v8h*)&kr[h * 16 + 8]);
        st0 = wmma_f16(a, Qb[ks], st0);
      }
      {
        const f16* kr = &kh[(size_t)(kblk + 16 + lm) * HD + ks * 32];
        v16h a = cat8(*(const v8h*)&kr[h * 16], *(const v8h*)&kr[h * 16 + 8]);
        st1 = wmma_f16(a, Qb[ks], st1);
      }
    }

    // Causal mask: key (row of S^T) > query (col) -> -inf.
    if (kblk + 31 > q0) {
      const int query = q0 + lm;
      #pragma unroll
      for (int r = 0; r < 8; ++r) {
        if (kblk + r + 8 * h > query)      st0[r] = -1e30f;
        if (kblk + 16 + r + 8 * h > query) st1[r] = -1e30f;
      }
    }

    // Per-query max (query = lane%16; reduce across the two half-wave copies).
    float mx = -1e30f;
    #pragma unroll
    for (int r = 0; r < 8; ++r) mx = fmaxf(mx, fmaxf(st0[r], st1[r]));
    mx = fmaxf(mx, __shfl_xor(mx, 16, 32));
    const float mnew  = fmaxf(mstat, mx);
    const float alpha = __expf(mstat - mnew);

    // Exponentiate; pack directly into the P*V A-fragment (layout matches!).
    float rs = 0.f;
    v16h pa;
    #pragma unroll
    for (int r = 0; r < 8; ++r) {
      float p0 = __expf(st0[r] - mnew);
      float p1 = __expf(st1[r] - mnew);
      rs += p0 + p1;
      pa[r]     = (f16)p0;
      pa[r + 8] = (f16)p1;
    }
    rs += __shfl_xor(rs, 16, 32);
    lstat = lstat * alpha + rs;
    mstat = mnew;

    // Rescale O: output row (query) of reg r is r + 8*h -> fetch that query's alpha.
    float ar[8];
    #pragma unroll
    for (int r = 0; r < 8; ++r) ar[r] = __shfl(alpha, r + 8 * h, 32);

    #pragma unroll
    for (int dt = 0; dt < 4; ++dt) {
      #pragma unroll
      for (int r = 0; r < 8; ++r) o[dt][r] *= ar[r];
      // V B-fragment: lane holds d = dt*16 + lane%16, elements key = kblk + h*8 + e (e<8), kblk + 16 + h*8 + (e-8) (e>=8).
      v16h vbf = cat8(*(const v8h*)&vh[(size_t)(dt * 16 + lm) * TT + kblk + h * 8],
                      *(const v8h*)&vh[(size_t)(dt * 16 + lm) * TT + kblk + 16 + h * 8]);
      o[dt] = wmma_f16(pa, vbf, o[dt]);
    }
  }

  // Finalize: divide by per-query normalizer, write concatenated-heads layout.
  float lr[8];
  #pragma unroll
  for (int r = 0; r < 8; ++r) lr[r] = __shfl(lstat, r + 8 * h, 32);
  #pragma unroll
  for (int dt = 0; dt < 4; ++dt) {
    #pragma unroll
    for (int r = 0; r < 8; ++r) {
      int t = q0 + r + 8 * h;
      att[((size_t)b * TT + t) * NE + head * HD + dt * 16 + lm] =
          (f16)(o[dt][r] / lr[r]);
    }
  }
}

// ---------------------------------------------------------------------------
// Kernel 3: output projection att[8192,768] (f16) x Wp[768,768] + bp -> fp32.
// Tile 256x64, K-step 32, A staged via async global->LDS, Wp transposed+cvt.
// ---------------------------------------------------------------------------
__global__ __launch_bounds__(256) void proj_kernel(
    const f16* __restrict__ att,
    const float* __restrict__ Wp,
    const float* __restrict__ bp,
    float* __restrict__ out)
{
  __shared__ f16 xt[256 * 32];
  __shared__ f16 wt[64 * 32];

  const int tid  = threadIdx.x;
  const int m0   = blockIdx.x * 256;
  const int n0   = blockIdx.y * 64;
  const int wave = tid >> 5;
  const int lane = tid & 31;
  const int lm   = lane & 15;
  const int h    = lane >> 4;

  v8f acc[2][4];
  #pragma unroll
  for (int s = 0; s < 2; ++s)
    #pragma unroll
    for (int i = 0; i < 4; ++i) acc[s][i] = vzero();

  for (int kc = 0; kc < NE; kc += 32) {
    // Stage A tile (f16) via async 16B copies.
    #pragma unroll
    for (int it = 0; it < 4; ++it) {
      int f   = tid + it * 256;       // 0..1023
      int row = f >> 2;
      int c8  = (f & 3) << 3;
      cp16_to_lds(&xt[row * 32 + c8], &att[(size_t)(m0 + row) * NE + kc + c8]);
    }
    // Stage Wp 32x64 transposed (fp32 -> f16).
    #pragma unroll
    for (int it = 0; it < 2; ++it) {
      int f  = tid + it * 256;        // 0..511
      int c  = f >> 4;
      int d4 = (f & 15) << 2;
      float4 v = *(const float4*)&Wp[(size_t)(kc + c) * NE + n0 + d4];
      wt[(d4 + 0) * 32 + c] = (f16)v.x;
      wt[(d4 + 1) * 32 + c] = (f16)v.y;
      wt[(d4 + 2) * 32 + c] = (f16)v.z;
      wt[(d4 + 3) * 32 + c] = (f16)v.w;
    }
    cp_async_wait();
    __syncthreads();

    const int r0 = wave * 32 + lm;
    v16h a0 = cat8(*(const v8h*)&xt[r0 * 32 + h * 16],
                   *(const v8h*)&xt[r0 * 32 + h * 16 + 8]);
    v16h a1 = cat8(*(const v8h*)&xt[(r0 + 16) * 32 + h * 16],
                   *(const v8h*)&xt[(r0 + 16) * 32 + h * 16 + 8]);
    #pragma unroll
    for (int nt = 0; nt < 4; ++nt) {
      v16h bf = *(const v16h*)&wt[(nt * 16 + lm) * 32 + h * 16];
      acc[0][nt] = wmma_f16(a0, bf, acc[0][nt]);
      acc[1][nt] = wmma_f16(a1, bf, acc[1][nt]);
    }
    __syncthreads();
  }

  #pragma unroll
  for (int nt = 0; nt < 4; ++nt) {
    float bias = bp[n0 + nt * 16 + lm];
    #pragma unroll
    for (int s = 0; s < 2; ++s) {
      #pragma unroll
      for (int r = 0; r < 8; ++r) {
        int g = m0 + wave * 32 + s * 16 + r + 8 * h;
        out[(size_t)g * NE + n0 + nt * 16 + lm] = acc[s][nt][r] + bias;
      }
    }
  }
}

// ---------------------------------------------------------------------------
extern "C" void kernel_launch(void* const* d_in, const int* in_sizes, int n_in,
                              void* d_out, int out_size, void* d_ws, size_t ws_size,
                              hipStream_t stream) {
  (void)in_sizes; (void)n_in; (void)out_size; (void)ws_size;

  const float* x  = (const float*)d_in[0];
  const float* Wq = (const float*)d_in[1];
  const float* Wk = (const float*)d_in[2];
  const float* Wv = (const float*)d_in[3];
  const float* Wp = (const float*)d_in[4];
  const float* bp = (const float*)d_in[5];
  float* out = (float*)d_out;

  const size_t perTensor = (size_t)BB * NH * TT * HD;  // 6,291,456 elems (== B*T*NE)
  f16* xh    = (f16*)d_ws;
  f16* qbuf  = xh + perTensor;
  f16* kbuf  = qbuf + perTensor;
  f16* vtbuf = kbuf + perTensor;
  f16* attc  = vtbuf + perTensor;

  // 0) x fp32 -> f16 once (3072 blocks x 256 threads x 8 elems).
  cvt_x_kernel<<<dim3(3072), 256, 0, stream>>>(x, xh);
  // 1) QKV projections: 32 M-tiles x (12 heads x {q,k,v}).
  qkv_proj_kernel<<<dim3(32, NH * 3), 256, 0, stream>>>(xh, Wq, Wk, Wv, qbuf, kbuf, vtbuf);
  // 2) Causal flash attention: 16 query tiles x 12 heads x 4 batches.
  attn_kernel<<<dim3(TT / 128, NH, BB), 256, 0, stream>>>(qbuf, kbuf, vtbuf, attc);
  // 3) Output projection: 32 M-tiles x 12 N-tiles.
  proj_kernel<<<dim3(32, NE / 64), 256, 0, stream>>>(attc, Wp, bp, out);
}
